// LGTAdapter_60808146977029
// MI455X (gfx1250) — matrix operand order repaired and checked by reading, not compiled, not run on hardware
//
#include <hip/hip_runtime.h>

// ---------------------------------------------------------------------------
// Problem constants (match reference)
// ---------------------------------------------------------------------------
static constexpr int B_  = 32;
static constexpr int N_  = 512;
static constexpr int D_  = 512;
static constexpr int H_  = 8;
static constexpr int HD_ = 64;
static constexpr int NL_ = 2;
static constexpr int DH_ = 256;

typedef __bf16 bf16;
typedef __attribute__((ext_vector_type(16))) __bf16 v16bf;
typedef __attribute__((ext_vector_type(8)))  float  v8f;

#if __has_builtin(__builtin_amdgcn_global_load_async_to_lds_b128) && \
    __has_builtin(__builtin_amdgcn_s_wait_asynccnt)
#define USE_ASYNC_LDS 1
// b128 async builtin is typed: (v4i AS1*, v4i AS3*, imm offset, imm cpol)
typedef int asv4i __attribute__((vector_size(16)));
typedef __attribute__((address_space(1))) asv4i gas_v4i;
typedef __attribute__((address_space(3))) asv4i lds_v4i;
#else
#define USE_ASYNC_LDS 0
#endif

// ---------------------------------------------------------------------------
// WMMA fragment helpers (CDNA5 16x16x32 bf16 layouts, CDNA5 ISA)
// A 16x32: lanes 0-15 -> M=lane, K in {0..7, 16..23}; lanes 16-31 -> M=lane-16,
//          K in {8..15, 24..31}.
// B 32x16: lane n -> column n (= row n of W for C = A*W^T), K same as A:
//          half 0 K in {0..7, 16..23}, half 1 K in {8..15, 24..31}.
// C/D 16x16: VGPR i -> row = half*8 + i, col = lane&15.
// ---------------------------------------------------------------------------
static __device__ __forceinline__ v16bf load_a_frag(const bf16* __restrict__ A, int lda,
                                                    int row, int kb, int m, int half) {
  const bf16* p0 = A + (size_t)(row + m) * lda + kb + half * 8;
  const bf16* p1 = p0 + 16;
  v16bf a;
#pragma unroll
  for (int j = 0; j < 8; ++j) { a[j] = p0[j]; a[8 + j] = p1[j]; }
  return a;
}

static __device__ __forceinline__ v16bf load_b_frag(const bf16* __restrict__ Bm, int ldb,
                                                    int col, int kb, int n, int half) {
  const bf16* p0 = Bm + (size_t)(col + n) * ldb + kb + half * 8;
  const bf16* p1 = p0 + 16;
  v16bf b;
#pragma unroll
  for (int j = 0; j < 8; ++j) { b[j] = p0[j]; b[8 + j] = p1[j]; }
  return b;
}

static __device__ __forceinline__ float qgelu_f(float v) {
  return v / (1.f + __expf(-1.702f * v));
}

// ---------------------------------------------------------------------------
// Generic WMMA GEMM:  C[bz] = act( A[bz] (MxK) * B[bz]^T (KxN) + bias + resid )
// A, B bf16 row-major; acc f32. One wave computes a 32x32 output tile
// (2x2 fragments -> 4 wmma per 32-deep K step, 2x arithmetic intensity vs
// 16x16). grid = (N/32, M/32, batch).
// ---------------------------------------------------------------------------
template <int ACT, bool OUTT>
__global__ __launch_bounds__(32) void k_gemm(const bf16* __restrict__ A, long long sA, int lda,
                                             const bf16* __restrict__ Bm, long long sB, int ldb,
                                             const float* __restrict__ bias,
                                             const float* __restrict__ resid, long long sR, int ldr,
                                             float* __restrict__ outF, long long sOF, int ldof,
                                             bf16* __restrict__ outB, long long sOB, int ldob,
                                             int K) {
  const int lane = threadIdx.x & 31;
  const int half = lane >> 4;
  const int mn   = lane & 15;
  const int col  = blockIdx.x * 32;
  const int row  = blockIdx.y * 32;
  const int bz   = blockIdx.z;

  const bf16* Ab = A + (size_t)bz * sA;
  const bf16* Bb = Bm + (size_t)bz * sB;

  v8f acc[2][2] = {};
  for (int kb = 0; kb < K; kb += 32) {
    if (kb + 32 < K) {  // prefetch next K tile (global_prefetch_b8)
      __builtin_prefetch(Ab + (size_t)(row + mn) * lda + kb + 32, 0, 0);
      __builtin_prefetch(Bb + (size_t)(col + mn) * ldb + kb + 32, 0, 0);
    }
    v16bf a0 = load_a_frag(Ab, lda, row,      kb, mn, half);
    v16bf a1 = load_a_frag(Ab, lda, row + 16, kb, mn, half);
    v16bf b0 = load_b_frag(Bb, ldb, col,      kb, mn, half);
    v16bf b1 = load_b_frag(Bb, ldb, col + 16, kb, mn, half);
    acc[0][0] = __builtin_amdgcn_wmma_f32_16x16x32_bf16(false, a0, false, b0, (short)0, acc[0][0], false, false);
    acc[0][1] = __builtin_amdgcn_wmma_f32_16x16x32_bf16(false, a0, false, b1, (short)0, acc[0][1], false, false);
    acc[1][0] = __builtin_amdgcn_wmma_f32_16x16x32_bf16(false, a1, false, b0, (short)0, acc[1][0], false, false);
    acc[1][1] = __builtin_amdgcn_wmma_f32_16x16x32_bf16(false, a1, false, b1, (short)0, acc[1][1], false, false);
  }

#pragma unroll
  for (int ni = 0; ni < 2; ++ni) {
    const int c  = col + ni * 16 + mn;
    const float bv = bias ? bias[c] : 0.f;
#pragma unroll
    for (int mi = 0; mi < 2; ++mi) {
#pragma unroll
      for (int i = 0; i < 8; ++i) {
        const int r = row + mi * 16 + half * 8 + i;
        float v = acc[mi][ni][i] + bv;
        if (resid) v += resid[(size_t)bz * sR + (size_t)r * ldr + c];
        if (ACT) v = qgelu_f(v);
        if (outF) outF[(size_t)bz * sOF + (size_t)r * ldof + c] = v;
        if (outB) {
          if (OUTT) outB[(size_t)bz * sOB + (size_t)c * ldob + r] = (bf16)v;
          else      outB[(size_t)bz * sOB + (size_t)r * ldob + c] = (bf16)v;
        }
      }
    }
  }
}

// ---------------------------------------------------------------------------
// Fused attention: one wave handles 16 query rows for one (b,h).
// The 16x512 f32 mask tile is staged into LDS with GLOBAL_LOAD_ASYNC_TO_LDS
// (ASYNCcnt) overlapped with the q k^T WMMA loop; scale+mask is fused into
// the softmax pass. Then o = p v with one prob-fragment feeding 4 WMMAs.
// grid = (N/16, B*H), block = 32.
// ---------------------------------------------------------------------------
__global__ __launch_bounds__(32) void k_attn(const bf16* __restrict__ q,
                                             const bf16* __restrict__ k,
                                             const bf16* __restrict__ vT,
                                             const float* __restrict__ mask,
                                             float* __restrict__ o) {
  __shared__ float sp[16][N_];      // 32 KB raw-score / prob tile
  __shared__ float smask[16][N_];   // 32 KB mask tile

  const int lane = threadIdx.x & 31;
  const int half = lane >> 4;
  const int mn   = lane & 15;
  const int qt   = blockIdx.x;
  const int bh   = blockIdx.y;

  // ---- kick off async mask tile copy (overlaps with score WMMAs) ----
  {
    const char* gsrc = (const char*)(mask + (size_t)(qt * 16) * N_);
    char* ldst = (char*)&smask[0][0];
#if USE_ASYNC_LDS
#pragma unroll 4
    for (int t = lane; t < 16 * N_ * 4 / 16; t += 32) {   // 2048 x 16B chunks
      __builtin_amdgcn_global_load_async_to_lds_b128(
          (gas_v4i*)(gsrc + (size_t)t * 16), (lds_v4i*)(ldst + (size_t)t * 16), 0, 0);
    }
#else
    const float* gs = (const float*)gsrc;
    float* ls = (float*)ldst;
    for (int t = lane; t < 16 * N_; t += 32) ls[t] = gs[t];
#endif
  }

  const bf16* qb = q  + (size_t)bh * N_ * HD_;
  const bf16* kb = k  + (size_t)bh * N_ * HD_;
  const bf16* vb = vT + (size_t)bh * HD_ * N_;

  // q fragments for full K=64 (two 32-chunks), loaded once.
  v16bf aq[2];
#pragma unroll
  for (int kk = 0; kk < 2; ++kk) aq[kk] = load_a_frag(qb, HD_, qt * 16, kk * 32, mn, half);

  // ---- raw scores = q k^T ----
  for (int kt = 0; kt < N_ / 16; ++kt) {
    v8f acc = {};
#pragma unroll
    for (int kk = 0; kk < 2; ++kk) {
      v16bf bfrg = load_b_frag(kb, HD_, kt * 16, kk * 32, mn, half);
      acc = __builtin_amdgcn_wmma_f32_16x16x32_bf16(false, aq[kk], false, bfrg, (short)0, acc, false, false);
    }
#pragma unroll
    for (int i = 0; i < 8; ++i) sp[half * 8 + i][kt * 16 + mn] = acc[i];
  }

#if USE_ASYNC_LDS
  __builtin_amdgcn_s_wait_asynccnt(0);
#endif
  __syncthreads();

  // ---- softmax per row (scale + mask fused); lane pair splits 512 cols ----
  {
    const float scale = 0.125f;  // 1/sqrt(64)
    const int r  = mn;
    const int c0 = half * 256;
    float mx = -3.0e38f;
    for (int c = 0; c < 256; ++c)
      mx = fmaxf(mx, sp[r][c0 + c] * scale + smask[r][c0 + c]);
    mx = fmaxf(mx, __shfl_xor(mx, 16, 32));
    float s = 0.f;
    for (int c = 0; c < 256; ++c) {
      float e = __expf(sp[r][c0 + c] * scale + smask[r][c0 + c] - mx);
      sp[r][c0 + c] = e;
      s += e;
    }
    s += __shfl_xor(s, 16, 32);
    const float inv = 1.f / s;
    for (int c = 0; c < 256; ++c) sp[r][c0 + c] *= inv;
  }
  __syncthreads();

  // ---- o = p v : one prob fragment feeds all 4 output tiles ----
  v8f oacc[HD_ / 16] = {};
  for (int kb2 = 0; kb2 < N_; kb2 += 32) {
    v16bf ap;
#pragma unroll
    for (int j = 0; j < 8; ++j) {
      ap[j]     = (bf16)sp[mn][kb2 + half * 8 + j];
      ap[8 + j] = (bf16)sp[mn][kb2 + 16 + half * 8 + j];
    }
#pragma unroll
    for (int ct = 0; ct < HD_ / 16; ++ct) {
      v16bf bfrg = load_b_frag(vb, N_, ct * 16, kb2, mn, half);
      oacc[ct] = __builtin_amdgcn_wmma_f32_16x16x32_bf16(false, ap, false, bfrg, (short)0, oacc[ct], false, false);
    }
  }
#pragma unroll
  for (int ct = 0; ct < HD_ / 16; ++ct)
#pragma unroll
    for (int i = 0; i < 8; ++i)
      o[((size_t)bh * N_ + qt * 16 + half * 8 + i) * HD_ + ct * 16 + mn] = oacc[ct][i];
}

// ---------------------------------------------------------------------------
// Small elementwise / reduction kernels
// ---------------------------------------------------------------------------
__global__ void k_cvt(const float* __restrict__ in, bf16* __restrict__ out, int n) {
  int i = blockIdx.x * 256 + threadIdx.x;
  if (i < n) out[i] = (bf16)in[i];
}

__global__ void k_tcvt(const float* __restrict__ in, bf16* __restrict__ out, int R, int C, int n) {
  int i = blockIdx.x * 256 + threadIdx.x;
  if (i < n) {
    int r = i / C, c = i % C;
    out[(size_t)c * R + r] = (bf16)in[i];
  }
}

__global__ void k_addpos(const float* __restrict__ hid, const float* __restrict__ pos,
                         float* __restrict__ x, int n) {
  int i = blockIdx.x * 256 + threadIdx.x;
  if (i < n) x[i] = hid[i] + pos[i % (N_ * D_)];
}

__global__ void k_ln(const float* __restrict__ x, const float* __restrict__ g,
                     const float* __restrict__ bta, bf16* __restrict__ out) {
  __shared__ float red[128];
  const int row = blockIdx.x, tid = threadIdx.x;
  const float* xr = x + (size_t)row * D_;
  float s = 0.f;
  for (int j = tid; j < D_; j += 128) s += xr[j];
  red[tid] = s; __syncthreads();
  for (int w = 64; w > 0; w >>= 1) { if (tid < w) red[tid] += red[tid + w]; __syncthreads(); }
  const float mu = red[0] / D_;
  __syncthreads();
  float v = 0.f;
  for (int j = tid; j < D_; j += 128) { float d = xr[j] - mu; v += d * d; }
  red[tid] = v; __syncthreads();
  for (int w = 64; w > 0; w >>= 1) { if (tid < w) red[tid] += red[tid + w]; __syncthreads(); }
  const float rstd = rsqrtf(red[0] / D_ + 1e-5f);
  bf16* orow = out + (size_t)row * D_;
  for (int j = tid; j < D_; j += 128) orow[j] = (bf16)((xr[j] - mu) * rstd * g[j] + bta[j]);
}

__global__ void k_norm(const float* __restrict__ x, float* __restrict__ nrm) {
  __shared__ float red[128];
  const int row = blockIdx.x, tid = threadIdx.x;
  const float* xr = x + (size_t)row * D_;
  float s = 0.f;
  for (int j = tid; j < D_; j += 128) s += xr[j] * xr[j];
  red[tid] = s; __syncthreads();
  for (int w = 64; w > 0; w >>= 1) { if (tid < w) red[tid] += red[tid + w]; __syncthreads(); }
  if (tid == 0) nrm[row] = sqrtf(red[0]);
}

__global__ void k_qkv_split(const float* __restrict__ qkv, bf16* __restrict__ q,
                            bf16* __restrict__ k, bf16* __restrict__ vT, int n) {
  int i = blockIdx.x * 256 + threadIdx.x;
  if (i >= n) return;
  const int d  = i % HD_;
  const int nn = (i / HD_) % N_;
  const int h  = (i / (HD_ * N_)) % H_;
  const int b  = i / (HD_ * N_ * H_);
  const size_t base = ((size_t)b * N_ + nn) * (3 * D_);
  const size_t bh = (size_t)b * H_ + h;
  q [(bh * N_ + nn) * HD_ + d] = (bf16)qkv[base + h * HD_ + d];
  k [(bh * N_ + nn) * HD_ + d] = (bf16)qkv[base + D_ + h * HD_ + d];
  vT[(bh * HD_ + d) * N_ + nn] = (bf16)qkv[base + 2 * D_ + h * HD_ + d];
}

__global__ void k_oresh(const float* __restrict__ o, bf16* __restrict__ obf, int n) {
  int i = blockIdx.x * 256 + threadIdx.x;
  if (i >= n) return;
  const int inner = i % D_;
  const int h = inner / HD_, d = inner % HD_;
  const int nn = (i / D_) % N_;
  const int b  = i / (D_ * N_);
  obf[i] = (bf16)o[(((size_t)b * H_ + h) * N_ + nn) * HD_ + d];
}

__global__ void k_costhresh(float* __restrict__ thr, const float* __restrict__ nrm, int n) {
  int i = blockIdx.x * 256 + threadIdx.x;
  if (i >= n) return;
  const int j = i % N_;
  const int m = (i / N_) % N_;
  const int b = i / (N_ * N_);
  const float den = nrm[b * N_ + m] * nrm[b * N_ + j] + 1e-20f;
  const float s = thr[i] / den;
  thr[i] = (s > 0.7f) ? s : 0.f;
}

__global__ void k_adjsm(const float* __restrict__ thr, const int* __restrict__ lengths,
                        bf16* __restrict__ adjb) {
  __shared__ float red[128];
  const int b = blockIdx.x / N_, m = blockIdx.x % N_;
  const int tid = threadIdx.x;
  const int L = lengths[b];
  const float* row = thr + ((size_t)b * N_ + m) * N_;
  bf16* orow = adjb + ((size_t)b * N_ + m) * N_;
  if (m >= L) {
    for (int c = tid; c < N_; c += 128) orow[c] = (bf16)0.f;
    return;
  }
  float mx = -3.0e38f;
  for (int c = tid; c < L; c += 128) mx = fmaxf(mx, row[c]);
  red[tid] = mx; __syncthreads();
  for (int w = 64; w > 0; w >>= 1) { if (tid < w) red[tid] = fmaxf(red[tid], red[tid + w]); __syncthreads(); }
  mx = red[0];
  __syncthreads();
  float s = 0.f;
  for (int c = tid; c < L; c += 128) s += __expf(row[c] - mx);
  red[tid] = s; __syncthreads();
  for (int w = 64; w > 0; w >>= 1) { if (tid < w) red[tid] += red[tid + w]; __syncthreads(); }
  const float inv = 1.f / red[0];
  for (int c = tid; c < N_; c += 128)
    orow[c] = (c < L) ? (bf16)(__expf(row[c] - mx) * inv) : (bf16)0.f;
}

__global__ void k_dadj(bf16* __restrict__ dadj, int n) {
  int i = blockIdx.x * 256 + threadIdx.x;
  if (i >= n) return;
  const int r = i / N_, c = i % N_;
  dadj[i] = (bf16)__expf(-fabsf((float)(r - c)) * (1.0f / 2.71828182845904523536f));
}

__global__ void k_concat(const float* __restrict__ x1, const float* __restrict__ x2,
                         bf16* __restrict__ ccat, int n) {
  int i = blockIdx.x * 256 + threadIdx.x;
  if (i >= n) return;
  const int j  = i % D_;
  const int nn = (i / D_) % N_;
  const int b  = i / (D_ * N_);
  const size_t rb = ((size_t)b * N_ + nn) * DH_;
  ccat[i] = (bf16)((j < DH_) ? x1[rb + j] : x2[rb + j - DH_]);
}

// ---------------------------------------------------------------------------
// Host-side launch helpers
// ---------------------------------------------------------------------------
static void launch_gemm(hipStream_t s, int act, int outT,
                        const bf16* A, long long sA, int lda,
                        const bf16* Bm, long long sB, int ldb,
                        const float* bias,
                        const float* resid, long long sR, int ldr,
                        float* outF, long long sOF, int ldof,
                        bf16* outB, long long sOB, int ldob,
                        int M, int N, int K, int batch) {
  dim3 g(N / 32, M / 32, batch), blk(32);
  if (!act && !outT)
    k_gemm<0, false><<<g, blk, 0, s>>>(A, sA, lda, Bm, sB, ldb, bias, resid, sR, ldr, outF, sOF, ldof, outB, sOB, ldob, K);
  else if (act && !outT)
    k_gemm<1, false><<<g, blk, 0, s>>>(A, sA, lda, Bm, sB, ldb, bias, resid, sR, ldr, outF, sOF, ldof, outB, sOB, ldob, K);
  else if (!act && outT)
    k_gemm<0, true><<<g, blk, 0, s>>>(A, sA, lda, Bm, sB, ldb, bias, resid, sR, ldr, outF, sOF, ldof, outB, sOB, ldob, K);
  else
    k_gemm<1, true><<<g, blk, 0, s>>>(A, sA, lda, Bm, sB, ldb, bias, resid, sR, ldr, outF, sOF, ldof, outB, sOB, ldob, K);
}

static inline dim3 ew_grid(int n) { return dim3((n + 255) / 256); }

extern "C" void kernel_launch(void* const* d_in, const int* in_sizes, int n_in,
                              void* d_out, int out_size, void* d_ws, size_t ws_size,
                              hipStream_t stream) {
  (void)in_sizes; (void)n_in; (void)out_size; (void)ws_size;

  const float* hidden = (const float*)d_in[0];
  const int*   lengths = (const int*)d_in[1];
  const float* mask   = (const float*)d_in[2];
  const float* pos    = (const float*)d_in[3];
  const float* ln1_g  = (const float*)d_in[4];
  const float* ln1_b  = (const float*)d_in[5];
  const float* in_w   = (const float*)d_in[6];
  const float* in_b   = (const float*)d_in[7];
  const float* out_w  = (const float*)d_in[8];
  const float* out_b  = (const float*)d_in[9];
  const float* ln2_g  = (const float*)d_in[10];
  const float* ln2_b  = (const float*)d_in[11];
  const float* fc_w   = (const float*)d_in[12];
  const float* fc_b   = (const float*)d_in[13];
  const float* pj_w   = (const float*)d_in[14];
  const float* pj_b   = (const float*)d_in[15];
  const float* gc1_w  = (const float*)d_in[16];
  const float* gc1_lw = (const float*)d_in[17];
  const float* gc1_lb = (const float*)d_in[18];
  const float* gc2_w  = (const float*)d_in[19];
  const float* gc3_w  = (const float*)d_in[20];
  const float* gc3_lw = (const float*)d_in[21];
  const float* gc3_lb = (const float*)d_in[22];
  const float* gc4_w  = (const float*)d_in[23];
  const float* lin_w  = (const float*)d_in[24];
  const float* lin_b  = (const float*)d_in[25];

  char* base = (char*)d_ws;
  size_t off = 0;
  auto alloc = [&](size_t bytes) -> char* {
    off = (off + 255) & ~(size_t)255;
    char* p = base + off;
    off += bytes;
    return p;
  };

  // ---- persistent buffers ----
  float* x    = (float*)alloc((size_t)B_ * N_ * D_ * 4);
  bf16*  hbf  = (bf16*)alloc((size_t)B_ * N_ * D_ * 2);   // LN out / later xbf

  // bf16 weights
  bf16* inw_bf  = (bf16*)alloc((size_t)NL_ * 3 * D_ * D_ * 2);
  bf16* outw_bf = (bf16*)alloc((size_t)NL_ * D_ * D_ * 2);
  bf16* fcw_bf  = (bf16*)alloc((size_t)NL_ * 4 * D_ * D_ * 2);
  bf16* pjw_bf  = (bf16*)alloc((size_t)NL_ * D_ * 4 * D_ * 2);
  bf16* g1lw_bf = (bf16*)alloc((size_t)DH_ * D_ * 2);
  bf16* g3lw_bf = (bf16*)alloc((size_t)DH_ * D_ * 2);
  bf16* linw_bf = (bf16*)alloc((size_t)D_ * D_ * 2);
  bf16* g1wT    = (bf16*)alloc((size_t)DH_ * D_ * 2);
  bf16* g2wT    = (bf16*)alloc((size_t)DH_ * DH_ * 2);
  bf16* g3wT    = (bf16*)alloc((size_t)DH_ * D_ * 2);
  bf16* g4wT    = (bf16*)alloc((size_t)DH_ * DH_ * 2);

  const size_t arena0 = off;

  // ---- weight conversions ----
  {
    int n;
    n = NL_ * 3 * D_ * D_; k_cvt<<<ew_grid(n), 256, 0, stream>>>(in_w,  inw_bf,  n);
    n = NL_ * D_ * D_;     k_cvt<<<ew_grid(n), 256, 0, stream>>>(out_w, outw_bf, n);
    n = NL_ * 4 * D_ * D_; k_cvt<<<ew_grid(n), 256, 0, stream>>>(fc_w,  fcw_bf,  n);
    n = NL_ * D_ * 4 * D_; k_cvt<<<ew_grid(n), 256, 0, stream>>>(pj_w,  pjw_bf,  n);
    n = DH_ * D_;          k_cvt<<<ew_grid(n), 256, 0, stream>>>(gc1_lw, g1lw_bf, n);
    n = DH_ * D_;          k_cvt<<<ew_grid(n), 256, 0, stream>>>(gc3_lw, g3lw_bf, n);
    n = D_ * D_;           k_cvt<<<ew_grid(n), 256, 0, stream>>>(lin_w,  linw_bf, n);
    n = D_ * DH_;          k_tcvt<<<ew_grid(n), 256, 0, stream>>>(gc1_w, g1wT, D_,  DH_, n);
    n = DH_ * DH_;         k_tcvt<<<ew_grid(n), 256, 0, stream>>>(gc2_w, g2wT, DH_, DH_, n);
    n = D_ * DH_;          k_tcvt<<<ew_grid(n), 256, 0, stream>>>(gc3_w, g3wT, D_,  DH_, n);
    n = DH_ * DH_;         k_tcvt<<<ew_grid(n), 256, 0, stream>>>(gc4_w, g4wT, DH_, DH_, n);
  }

  const int nBND = B_ * N_ * D_;     // 8.4M
  const int M    = B_ * N_;          // 16384 rows

  // x = hidden + pos
  k_addpos<<<ew_grid(nBND), 256, 0, stream>>>(hidden, pos, x, nBND);

  // ---- phase A arena ----
  float* qkv  = (float*)alloc((size_t)M * 3 * D_ * 4);
  bf16*  qbf  = (bf16*)alloc((size_t)B_ * H_ * N_ * HD_ * 2);
  bf16*  kbf  = (bf16*)alloc((size_t)B_ * H_ * N_ * HD_ * 2);
  bf16*  vT   = (bf16*)alloc((size_t)B_ * H_ * HD_ * N_ * 2);
  float* oatt = (float*)alloc((size_t)B_ * H_ * N_ * HD_ * 4);
  bf16*  obf  = (bf16*)alloc((size_t)M * D_ * 2);
  bf16*  fcbf = (bf16*)alloc((size_t)M * 4 * D_ * 2);

  for (int l = 0; l < NL_; ++l) {
    // LN1 -> hbf
    k_ln<<<dim3(M), 128, 0, stream>>>(x, ln1_g + l * D_, ln1_b + l * D_, hbf);
    // qkv = hbf @ in_w^T + in_b
    launch_gemm(stream, 0, 0,
                hbf, 0, D_, inw_bf + (size_t)l * 3 * D_ * D_, 0, D_,
                in_b + l * 3 * D_, nullptr, 0, 0,
                qkv, 0, 3 * D_, nullptr, 0, 0,
                M, 3 * D_, D_, 1);
    // split heads
    { int n = B_ * H_ * N_ * HD_;
      k_qkv_split<<<ew_grid(n), 256, 0, stream>>>(qkv, qbf, kbf, vT, n); }
    // attention
    k_attn<<<dim3(N_ / 16, B_ * H_), 32, 0, stream>>>(qbf, kbf, vT, mask, oatt);
    // heads -> [B,N,D] bf16
    k_oresh<<<ew_grid(nBND), 256, 0, stream>>>(oatt, obf, nBND);
    // x += obf @ out_w^T + out_b
    launch_gemm(stream, 0, 0,
                obf, 0, D_, outw_bf + (size_t)l * D_ * D_, 0, D_,
                out_b + l * D_, x, 0, D_,
                x, 0, D_, nullptr, 0, 0,
                M, D_, D_, 1);
    // LN2 -> hbf
    k_ln<<<dim3(M), 128, 0, stream>>>(x, ln2_g + l * D_, ln2_b + l * D_, hbf);
    // fc: qgelu(hbf @ fc_w^T + fc_b) -> fcbf (bf16)
    launch_gemm(stream, 1, 0,
                hbf, 0, D_, fcw_bf + (size_t)l * 4 * D_ * D_, 0, D_,
                fc_b + l * 4 * D_, nullptr, 0, 0,
                nullptr, 0, 0, fcbf, 0, 4 * D_,
                M, 4 * D_, D_, 1);
    // x += fcbf @ pj_w^T + pj_b
    launch_gemm(stream, 0, 0,
                fcbf, 0, 4 * D_, pjw_bf + (size_t)l * D_ * 4 * D_, 0, 4 * D_,
                pj_b + l * D_, x, 0, D_,
                x, 0, D_, nullptr, 0, 0,
                M, D_, 4 * D_, 1);
  }

  // ---- phase B: reuse arena ----
  off = arena0;
  float* thr    = (float*)alloc((size_t)B_ * N_ * N_ * 4);
  bf16*  adjbf  = (bf16*)alloc((size_t)B_ * N_ * N_ * 2);
  bf16*  dadjbf = (bf16*)alloc((size_t)N_ * N_ * 2);
  bf16*  xw1T   = (bf16*)alloc((size_t)B_ * DH_ * N_ * 2);
  float* lin1   = (float*)alloc((size_t)M * DH_ * 4);
  float* x1h    = (float*)alloc((size_t)M * DH_ * 4);
  bf16*  x1hbf  = (bf16*)alloc((size_t)M * DH_ * 2);
  bf16*  t2T    = (bf16*)alloc((size_t)B_ * DH_ * N_ * 2);
  float* x1f    = (float*)alloc((size_t)M * DH_ * 4);
  bf16*  xw3T   = (bf16*)alloc((size_t)B_ * DH_ * N_ * 2);
  float* lin3   = (float*)alloc((size_t)M * DH_ * 4);
  float* x2h    = (float*)alloc((size_t)M * DH_ * 4);
  bf16*  x2hbf  = (bf16*)alloc((size_t)M * DH_ * 2);
  bf16*  t4T    = (bf16*)alloc((size_t)B_ * DH_ * N_ * 2);
  float* x2f    = (float*)alloc((size_t)M * DH_ * 4);
  bf16*  ccat   = (bf16*)alloc((size_t)M * D_ * 2);
  float* nrm    = (float*)alloc((size_t)M * 4);

  bf16* xbf = hbf;  // reuse persistent bf16 buffer for final x
  k_cvt<<<ew_grid(nBND), 256, 0, stream>>>(x, xbf, nBND);
  k_norm<<<dim3(M), 128, 0, stream>>>(x, nrm);

  // sim = x x^T (batched), then cosine-threshold, then masked row softmax
  launch_gemm(stream, 0, 0,
              xbf, (long long)N_ * D_, D_, xbf, (long long)N_ * D_, D_,
              nullptr, nullptr, 0, 0,
              thr, (long long)N_ * N_, N_, nullptr, 0, 0,
              N_, N_, D_, B_);
  { int n = B_ * N_ * N_;
    k_costhresh<<<ew_grid(n), 256, 0, stream>>>(thr, nrm, n); }
  k_adjsm<<<dim3(B_ * N_), 128, 0, stream>>>(thr, lengths, adjbf);
  { int n = N_ * N_;
    k_dadj<<<ew_grid(n), 256, 0, stream>>>(dadjbf, n); }

  // ---- GCN branch 1 (adj) ----
  launch_gemm(stream, 0, 1,
              xbf, (long long)N_ * D_, D_, g1wT, 0, D_,
              nullptr, nullptr, 0, 0,
              nullptr, 0, 0, xw1T, (long long)DH_ * N_, N_,
              N_, DH_, D_, B_);
  launch_gemm(stream, 0, 0,
              xbf, 0, D_, g1lw_bf, 0, D_,
              gc1_lb, nullptr, 0, 0,
              lin1, 0, DH_, nullptr, 0, 0,
              M, DH_, D_, 1);
  launch_gemm(stream, 1, 0,
              adjbf, (long long)N_ * N_, N_, xw1T, (long long)DH_ * N_, N_,
              nullptr, lin1, (long long)N_ * DH_, DH_,
              x1h, (long long)N_ * DH_, DH_, x1hbf, (long long)N_ * DH_, DH_,
              N_, DH_, N_, B_);
  launch_gemm(stream, 0, 1,
              x1hbf, (long long)N_ * DH_, DH_, g2wT, 0, DH_,
              nullptr, nullptr, 0, 0,
              nullptr, 0, 0, t2T, (long long)DH_ * N_, N_,
              N_, DH_, DH_, B_);
  launch_gemm(stream, 1, 0,
              adjbf, (long long)N_ * N_, N_, t2T, (long long)DH_ * N_, N_,
              nullptr, x1h, (long long)N_ * DH_, DH_,
              x1f, (long long)N_ * DH_, DH_, nullptr, 0, 0,
              N_, DH_, N_, B_);

  // ---- GCN branch 2 (distance adj, shared A: strideA=0) ----
  launch_gemm(stream, 0, 1,
              xbf, (long long)N_ * D_, D_, g3wT, 0, D_,
              nullptr, nullptr, 0, 0,
              nullptr, 0, 0, xw3T, (long long)DH_ * N_, N_,
              N_, DH_, D_, B_);
  launch_gemm(stream, 0, 0,
              xbf, 0, D_, g3lw_bf, 0, D_,
              gc3_lb, nullptr, 0, 0,
              lin3, 0, DH_, nullptr, 0, 0,
              M, DH_, D_, 1);
  launch_gemm(stream, 1, 0,
              dadjbf, 0, N_, xw3T, (long long)DH_ * N_, N_,
              nullptr, lin3, (long long)N_ * DH_, DH_,
              x2h, (long long)N_ * DH_, DH_, x2hbf, (long long)N_ * DH_, DH_,
              N_, DH_, N_, B_);
  launch_gemm(stream, 0, 1,
              x2hbf, (long long)N_ * DH_, DH_, g4wT, 0, DH_,
              nullptr, nullptr, 0, 0,
              nullptr, 0, 0, t4T, (long long)DH_ * N_, N_,
              N_, DH_, DH_, B_);
  launch_gemm(stream, 1, 0,
              dadjbf, 0, N_, t4T, (long long)DH_ * N_, N_,
              nullptr, x2h, (long long)N_ * DH_, DH_,
              x2f, (long long)N_ * DH_, DH_, nullptr, 0, 0,
              N_, DH_, N_, B_);

  // concat + final linear
  k_concat<<<ew_grid(nBND), 256, 0, stream>>>(x1f, x2f, ccat, nBND);
  launch_gemm(stream, 0, 0,
              ccat, 0, D_, linw_bf, 0, D_,
              lin_b, nullptr, 0, 0,
              (float*)d_out, 0, D_, nullptr, 0, 0,
              M, D_, D_, 1);
}
